// TokenSelectionModule_76441827934911
// MI455X (gfx1250) — matrix operand order repaired and checked by reading, not compiled, not run on hardware
//
#include <hip/hip_runtime.h>
#include <hip/hip_bf16.h>

typedef _Float16 v8h  __attribute__((ext_vector_type(8)));
typedef _Float16 v16h __attribute__((ext_vector_type(16)));
typedef float    v8f  __attribute__((ext_vector_type(8)));

#define Bn 8
#define Cn 64
#define Hn 256
#define Wn 256
#define TILE 16
#define HALO 18                  // TILE + 2 (SAME padding halo)
#define PIX  (HALO*HALO)         // 324 pixels per halo tile
#define XH_STRIDE 72             // padded pixel stride in halves (144B) -> fewer LDS bank conflicts

// ---- LDS layout (bytes). Total ~259 KB; CDNA5 allows up to 320 KB per workgroup.
#define OFF_XF0   0
#define OFF_XF1   (OFF_XF0 + Cn*PIX*4)                 // f32 tiles: exact values for selection
#define OFF_XH0   (OFF_XF1 + Cn*PIX*4)                 // f16 tiles, [row][col][ch] for WMMA A loads
#define OFF_XH1   (OFF_XH0 + PIX*XH_STRIDE*2)
#define OFF_WH0   (OFF_XH1 + PIX*XH_STRIDE*2)          // folded weights W'[tap][ch] f16
#define OFF_WH1   (OFF_WH0 + 9*Cn*2)
#define OFF_S0    (OFF_WH1 + 9*Cn*2)                   // per-pixel scores
#define OFF_S1    (OFF_S0 + TILE*TILE*4)
#define OFF_B0    (OFF_S1 + TILE*TILE*4)               // folded biases (2 floats)
#define OFF_RED   (OFF_B0 + 2*4)                       // count reduction scratch
#define LDS_BYTES (OFF_RED + 256*4)

extern "C" __global__ __launch_bounds__(256)
void tsm_main(const float* __restrict__ x0, const float* __restrict__ x1,
              const float* __restrict__ dww0, const float* __restrict__ dwb0,
              const float* __restrict__ pww0, const float* __restrict__ pwb0,
              const float* __restrict__ dww1, const float* __restrict__ dwb1,
              const float* __restrict__ pww1, const float* __restrict__ pwb1,
              float* __restrict__ out, unsigned int* __restrict__ gcount)
{
    extern __shared__ char smem[];
    // NOTE: no pointer arrays from `smem` (would emit addrspacecast static initializers).
    float*    xf0 = (float*)(smem + OFF_XF0);
    float*    xf1 = (float*)(smem + OFF_XF1);
    _Float16* xh0 = (_Float16*)(smem + OFF_XH0);
    _Float16* xh1 = (_Float16*)(smem + OFF_XH1);
    _Float16* wh0 = (_Float16*)(smem + OFF_WH0);
    _Float16* wh1 = (_Float16*)(smem + OFF_WH1);
    float*    s0a = (float*)(smem + OFF_S0);
    float*    s1a = (float*)(smem + OFF_S1);
    float*    b0s = (float*)(smem + OFF_B0);
    unsigned* red = (unsigned*)(smem + OFF_RED);

    const int tid = threadIdx.x;
    const int tx = blockIdx.x, ty = blockIdx.y, bb = blockIdx.z;

    // ---------- Phase 0: fold weights: W'[c,t] = pw_w[c]*dw_w[c,t]; b0 = pw_b + sum pw_w*dw_b
    for (int e = tid; e < 2*Cn*9; e += 256) {
        int i = e / (Cn*9);
        int j = e - i*(Cn*9);
        int c = j / 9, t = j - c*9;
        const float* dww = i ? dww1 : dww0;
        const float* pww = i ? pww1 : pww0;
        _Float16*    whp = i ? wh1  : wh0;
        whp[t*Cn + c] = (_Float16)(pww[c] * dww[j]);
    }
    if (tid < 2) {
        const float* pwb = tid ? pwb1 : pwb0;
        const float* dwb = tid ? dwb1 : dwb0;
        const float* pw  = tid ? pww1 : pww0;
        float s = pwb[0];
        for (int c = 0; c < Cn; ++c) s += pw[c] * dwb[c];
        b0s[tid] = s;
    }

    // ---------- Phase 1: stage both 18x18x64 halo tiles into LDS (f32 exact + f16 for WMMA)
    for (int i = 0; i < 2; ++i) {
        const float* src = (i ? x1 : x0) + (size_t)bb * Cn * Hn * Wn;
        float*       xfp = i ? xf1 : xf0;
        _Float16*    xhp = i ? xh1 : xh0;
        for (int c = 0; c < Cn; ++c) {
            const float* plane = src + (size_t)c * Hn * Wn;
            for (int p = tid; p < PIX; p += 256) {
                int ry = p / HALO, rx = p - ry*HALO;
                int gy = ty*TILE + ry - 1, gx = tx*TILE + rx - 1;
                float v = 0.f;
                if ((unsigned)gy < (unsigned)Hn && (unsigned)gx < (unsigned)Wn)
                    v = plane[gy*Wn + gx];
                xfp[c*PIX + p] = v;                  // exact value for selection
                xhp[p*XH_STRIDE + c] = (_Float16)v;  // channel-contiguous for WMMA A
            }
        }
    }
    __syncthreads();

    // ---------- Phase 2: conv-as-GEMM on WMMA.
    // score_pre[pixel m] = sum_{k=0..63, tap t} A_t[m,k] * W'[k,t];  per tap: D += A_t x B_t.
    // A (16x32 f16): lanes0-15 m=0..15 hold K octets {0..7,16..23}; lanes16-31 hold {8..15,24..31}.
    // B (32x16 f16): lanes0-15 hold K octets {0..7,16..23}; lanes16-31 hold {8..15,24..31} (same value along N -> broadcast w).
    {
        int w = tid >> 5, lane = tid & 31;
        int inp = w & 1;            // waves alternate between x0 / x1 branches
        int rb  = w >> 1;           // base output row; each wave does rows rb, rb+4, rb+8, rb+12
        const _Float16* X  = inp ? xh1 : xh0;
        const _Float16* Wt = inp ? wh1 : wh0;
        float* S           = inp ? s1a : s0a;
        float bias = b0s[inp];
        int m  = lane & 15;         // pixel column = WMMA M index
        int hi = lane >> 4;

        for (int rr = 0; rr < 4; ++rr) {
            int r = rb + rr*4;      // output row 0..15; uses input rows r..r+2, cols m..m+2
            v8f acc = {0.f,0.f,0.f,0.f,0.f,0.f,0.f,0.f};
            for (int kap = 0; kap < 2; ++kap) {          // channel chunks 0..31 / 32..63
                const _Float16* Abase = X  + kap*32 + hi*8;
                const _Float16* Bbase = Wt + kap*32 + hi*8;
                #pragma unroll
                for (int t = 0; t < 9; ++t) {
                    int dy = t / 3, dx = t - dy*3;
                    const _Float16* pA = Abase + ((r+dy)*HALO + (m+dx))*XH_STRIDE;
                    v8h alo = *(const v8h*)pA;           // channels k..k+7   (ds_load_b128)
                    v8h ahi = *(const v8h*)(pA + 16);    // channels k+16..k+23
                    v16h a = __builtin_shufflevector(alo, ahi,
                              0,1,2,3,4,5,6,7,8,9,10,11,12,13,14,15);
                    v8h blo = *(const v8h*)(Bbase + t*Cn);
                    v8h bhi = *(const v8h*)(Bbase + t*Cn + 16);
                    v16h b = __builtin_shufflevector(blo, bhi,
                              0,1,2,3,4,5,6,7,8,9,10,11,12,13,14,15);
                    acc = __builtin_amdgcn_wmma_f32_16x16x32_f16(
                              false, a, false, b, (short)0, acc, false, false);
                }
            }
            // D layout: lane n (0-15) holds M=j in vgpr j; lanes 16-31 hold M=8+j.
            // All N columns identical -> lanes 0 and 16 publish the 16 pixel sums.
            if (m == 0) {
                #pragma unroll
                for (int j = 0; j < 8; ++j) {
                    float v = acc[j] + bias;
                    S[r*TILE + hi*8 + j] = 1.f / (1.f + __expf(-v));   // sigmoid
                }
            }
        }
    }
    __syncthreads();

    // ---------- Phase 3: selection + store + count (reads only LDS, fully coalesced stores)
    {
        int py = tid >> 4, px = tid & 15;
        float s0v = s0a[tid] + 1.f;   // w = (s+1)*x
        float s1v = s1a[tid] + 1.f;
        int gy = ty*TILE + py, gx = tx*TILE + px;
        int lofs = (py+1)*HALO + (px+1);
        size_t obase = (((size_t)bb*Cn)*Hn + gy)*(size_t)Wn + gx;
        unsigned cnt = 0;
        for (int c = 0; c < Cn; ++c) {
            float a0 = xf0[c*PIX + lofs];
            float a1 = xf1[c*PIX + lofs];
            bool pick0 = (s0v*a0 >= s1v*a1);   // argmax tie -> index 0
            out[obase + (size_t)c*Hn*Wn] = pick0 ? a0 : a1;
            cnt += pick0 ? 1u : 0u;
        }
        red[tid] = cnt;
    }
    __syncthreads();
    for (int st = 128; st > 0; st >>= 1) {
        if (tid < st) red[tid] += red[tid + st];
        __syncthreads();
    }
    if (tid == 0) atomicAdd(gcount, red[0]);
}

extern "C" __global__ void tsm_init(unsigned* g) { if (threadIdx.x == 0) *g = 0u; }

extern "C" __global__ void tsm_final(const unsigned* __restrict__ g, float* __restrict__ out)
{
    if (threadIdx.x == 0) {
        const size_t N = (size_t)Bn * Cn * Hn * Wn;
        float p0 = (float)(*g) / (float)N;
        out[N]     = p0;
        out[N + 1] = 1.f - p0;
    }
}

extern "C" void kernel_launch(void* const* d_in, const int* in_sizes, int n_in,
                              void* d_out, int out_size, void* d_ws, size_t ws_size,
                              hipStream_t stream)
{
    const float* x0   = (const float*)d_in[0];
    const float* x1   = (const float*)d_in[1];
    const float* dww0 = (const float*)d_in[2];
    const float* dwb0 = (const float*)d_in[3];
    const float* pww0 = (const float*)d_in[4];
    const float* pwb0 = (const float*)d_in[5];
    const float* dww1 = (const float*)d_in[6];
    const float* dwb1 = (const float*)d_in[7];
    const float* pww1 = (const float*)d_in[8];
    const float* pwb1 = (const float*)d_in[9];
    float* out = (float*)d_out;
    unsigned* gcount = (unsigned*)d_ws;

    tsm_init<<<1, 1, 0, stream>>>(gcount);
    dim3 grid(Wn/TILE, Hn/TILE, Bn);   // 16 x 16 x 8 tiles
    tsm_main<<<grid, 256, LDS_BYTES, stream>>>(x0, x1, dww0, dwb0, pww0, pwb0,
                                               dww1, dwb1, pww1, pwb1, out, gcount);
    tsm_final<<<1, 1, 0, stream>>>(gcount, out);
}
